// Net_6983616823390
// MI455X (gfx1250) — 16-bit and 8-bit matrix operand order repaired and checked by reading, compiled, not run on hardware
//
#include <hip/hip_runtime.h>
#include <hip/hip_bf16.h>

typedef __attribute__((ext_vector_type(16))) __bf16 v16bf;
typedef __attribute__((ext_vector_type(8)))  __bf16 v8bf;
typedef __attribute__((ext_vector_type(4)))  __bf16 v4bf;
typedef __attribute__((ext_vector_type(8)))  float  v8f;
typedef __attribute__((ext_vector_type(8)))  int    v8i;

namespace {
constexpr int kB   = 8192;
constexpr int kDin = 3072;
constexpr int kH   = 2000;
constexpr int kHP  = 2048;
constexpr int kC   = 10;

constexpr size_t OFF_SCAL  = 0;
constexpr size_t OFF_W1S   = 256;
constexpr size_t SZ_W1S    = (size_t)kHP * kDin * 2;
constexpr size_t OFF_WS2   = OFF_W1S + SZ_W1S;
constexpr size_t SZ_WSGN   = (size_t)kHP * kHP;
constexpr size_t OFF_WS3   = OFF_WS2 + SZ_WSGN;
constexpr size_t OFF_WS4   = OFF_WS3 + SZ_WSGN;
constexpr size_t OFF_BITS0 = OFF_WS4 + SZ_WSGN;
constexpr size_t SZ_BITS   = (size_t)kB * kHP;
constexpr size_t OFF_BITS1 = OFF_BITS0 + SZ_BITS;
constexpr size_t OFF_H4    = OFF_BITS1 + SZ_BITS;
constexpr size_t SZ_H4     = (size_t)kB * kH * 4;
constexpr size_t OFF_XBF   = OFF_H4 + SZ_H4;
}

__device__ __forceinline__ void async_b128(unsigned ldsOff, unsigned gOff,
                                           const void* base) {
  asm volatile("global_load_async_to_lds_b128 %0, %1, %2"
               :: "v"(ldsOff), "v"(gOff), "s"(base) : "memory");
}
__device__ __forceinline__ void wait_async0() {
  asm volatile("s_wait_asynccnt 0x0" ::: "memory");
}
__device__ __forceinline__ unsigned lds_off(const void* p) {
  return (unsigned)(unsigned long long)p;
}

__global__ void k_zero_scal(float* s) {
  if (threadIdx.x < 16) s[threadIdx.x] = 0.0f;
}

__global__ void k_abssum(const float* __restrict__ w, int n, float* __restrict__ dst) {
  float p = 0.0f;
  for (int i = blockIdx.x * blockDim.x + threadIdx.x; i < n; i += gridDim.x * blockDim.x)
    p += fabsf(w[i]);
  for (int off = 16; off; off >>= 1) p += __shfl_down(p, off, 32);
  if ((threadIdx.x & 31) == 0) atomicAdd(dst, p);
}

__global__ void k_x_bf16(const float* __restrict__ x, __bf16* __restrict__ o) {
  const int total = kB * kDin / 4;
  for (int i = blockIdx.x * blockDim.x + threadIdx.x; i < total; i += gridDim.x * blockDim.x) {
    const float4 f = ((const float4*)x)[i];
    v4bf o4;
    o4[0] = (__bf16)f.x; o4[1] = (__bf16)f.y; o4[2] = (__bf16)f.z; o4[3] = (__bf16)f.w;
    ((v4bf*)o)[i] = o4;
  }
}

__global__ void k_sign_bf16(const float* __restrict__ w, __bf16* __restrict__ o) {
  const int total = kHP * kDin;
  for (int i = blockIdx.x * blockDim.x + threadIdx.x; i < total; i += gridDim.x * blockDim.x) {
    const int n = i / kDin;
    float s = 0.0f;
    if (n < kH) {
      const float v = w[(size_t)n * kDin + (i % kDin)];
      s = (v > 0.0f) ? 1.0f : ((v < 0.0f) ? -1.0f : 0.0f);
    }
    o[i] = (__bf16)s;
  }
}

__global__ void k_sign_i8(const float* __restrict__ w, signed char* __restrict__ o) {
  const int total = kHP * kHP;
  for (int i = blockIdx.x * blockDim.x + threadIdx.x; i < total; i += gridDim.x * blockDim.x) {
    const int n = i >> 11;
    const int k = i & (kHP - 1);
    signed char s = 0;
    if (n < kH && k < kH) {
      const float v = w[(size_t)n * kH + k];
      s = (v > 0.0f) ? 1 : ((v < 0.0f) ? -1 : 0);
    }
    o[i] = s;
  }
}

__global__ __launch_bounds__(128) void k_gemm_l1(
    const __bf16* __restrict__ xbf, const __bf16* __restrict__ w1s,
    const float* __restrict__ b1, const float* __restrict__ sumW,
    unsigned char* __restrict__ bitsOut, float* __restrict__ sumH)
{
  __shared__ __align__(16) unsigned char sB[2][64 * 80];

  const int lane  = threadIdx.x;
  const int tid   = threadIdx.y * 32 + lane;
  const int lh    = lane >> 4;
  const int lm    = lane & 15;
  const int tileM = (blockIdx.x * 4 + threadIdx.y) * 16;
  const int tileN = blockIdx.y * 64;
  const __bf16* xrow = xbf + (size_t)(tileM + lm) * kDin;

  const int srow  = tid >> 1;
  const int shalf = (tid & 1) * 32;
  const unsigned gRow   = (unsigned)((tileN + srow) * (kDin * 2) + shalf);
  const unsigned ldsRow = (unsigned)(srow * 80 + shalf);
  const unsigned ldsB0  = lds_off(&sB[0][0]);

  v8f acc0, acc1, acc2, acc3;
  #pragma unroll
  for (int r = 0; r < 8; ++r) { acc0[r] = 0.0f; acc1[r] = 0.0f; acc2[r] = 0.0f; acc3[r] = 0.0f; }

  #define L1_LDB(P) __builtin_shufflevector(*(const v8bf*)(P), *(const v8bf*)((P) + 16), \
                                            0, 1, 2, 3, 4, 5, 6, 7, 8, 9, 10, 11, 12, 13, 14, 15)
  #define L1_COMPUTE(KS, BUF)                                                            \
    {                                                                                    \
      const int ka0 = (KS) * 32 + lh * 16;                                               \
      const v8bf a0 = *(const v8bf*)(xrow + ka0);                                        \
      const v8bf a1 = *(const v8bf*)(xrow + ka0 + 8);                                    \
      const v16bf a = __builtin_shufflevector(a0, a1, 0, 1, 2, 3, 4, 5, 6, 7,            \
                                              8, 9, 10, 11, 12, 13, 14, 15);             \
      const unsigned char* p = &sB[BUF][0] + lm * 80 + lh * 32;                          \
      const v16bf b0 = L1_LDB(p);                                                        \
      const v16bf b1 = L1_LDB(p + 16 * 80);                                              \
      const v16bf b2 = L1_LDB(p + 32 * 80);                                              \
      const v16bf b3 = L1_LDB(p + 48 * 80);                                              \
      acc0 = __builtin_amdgcn_wmma_f32_16x16x32_bf16(false, a, false, b0, (short)0, acc0, false, false); \
      acc1 = __builtin_amdgcn_wmma_f32_16x16x32_bf16(false, a, false, b1, (short)0, acc1, false, false); \
      acc2 = __builtin_amdgcn_wmma_f32_16x16x32_bf16(false, a, false, b2, (short)0, acc2, false, false); \
      acc3 = __builtin_amdgcn_wmma_f32_16x16x32_bf16(false, a, false, b3, (short)0, acc3, false, false); \
    }

  async_b128(ldsB0 + ldsRow,      gRow,      w1s);
  async_b128(ldsB0 + ldsRow + 16, gRow + 16, w1s);

  constexpr int NS = kDin / 32;
  for (int ks = 0; ks < NS; ks += 2) {
    wait_async0();
    __syncthreads();
    {
      const unsigned l = ldsB0 + 64u * 80u + ldsRow;
      const unsigned g = gRow + (unsigned)((ks + 1) * 64);
      async_b128(l,      g,      w1s);
      async_b128(l + 16, g + 16, w1s);
    }
    L1_COMPUTE(ks, 0)
    wait_async0();
    __syncthreads();
    if (ks + 2 < NS) {
      const unsigned g = gRow + (unsigned)((ks + 2) * 64);
      async_b128(ldsB0 + ldsRow,      g,      w1s);
      async_b128(ldsB0 + ldsRow + 16, g + 16, w1s);
    }
    L1_COMPUTE(ks + 1, 1)
  }
  #undef L1_COMPUTE
  #undef L1_LDB

  const float alpha = sumW[0] * (1.0f / ((float)kH * (float)kDin));
  const v8f accs[4] = {acc0, acc1, acc2, acc3};
  float lsum = 0.0f;
  #pragma unroll
  for (int t = 0; t < 4; ++t) {
    const int n = tileN + t * 16 + lm;
    const float bias = (n < kH) ? b1[n] : 0.0f;
    #pragma unroll
    for (int r = 0; r < 8; ++r) {
      const int m = tileM + lh * 8 + r;
      float v = alpha * accs[t][r] + bias;
      v = (v > 0.0f) ? v : 0.0f;
      bitsOut[(size_t)m * kHP + n] = (n < kH && v > 0.0f) ? 1 : 0;
      if (n < kH) lsum += v;
    }
  }
  for (int off = 16; off; off >>= 1) lsum += __shfl_down(lsum, off, 32);
  if (lane == 0) atomicAdd(sumH, lsum);
}

__global__ __launch_bounds__(128) void k_gemm_bin(
    const unsigned char* __restrict__ bitsIn, const signed char* __restrict__ wsgn,
    const float* __restrict__ bias, const float* __restrict__ sumW, float invW,
    const float* __restrict__ sumHin, float invH,
    unsigned char* __restrict__ bitsOut, float* __restrict__ sumHout,
    float* __restrict__ h4out)
{
  __shared__ __align__(16) unsigned char sB[2][64 * 80];

  const int lane  = threadIdx.x;
  const int tid   = threadIdx.y * 32 + lane;
  const int lh    = lane >> 4;
  const int lm    = lane & 15;
  const int tileM = (blockIdx.x * 4 + threadIdx.y) * 16;
  const int tileN = blockIdx.y * 64;
  const unsigned char* arow = bitsIn + (size_t)(tileM + lm) * kHP;

  const int srow  = tid >> 1;
  const int shalf = (tid & 1) * 32;
  const unsigned gRow   = (unsigned)((tileN + srow) * kHP + shalf);
  const unsigned ldsRow = (unsigned)(srow * 80 + shalf);
  const unsigned ldsB0  = lds_off(&sB[0][0]);

  v8i acc0, acc1, acc2, acc3;
  #pragma unroll
  for (int r = 0; r < 8; ++r) { acc0[r] = 0; acc1[r] = 0; acc2[r] = 0; acc3[r] = 0; }

  #define BIN_LDB(P)                                                      \
    ({ const int4 d0_ = *(const int4*)(P);                                \
       const int4 d1_ = *(const int4*)((P) + 32);                         \
       v8i b_; b_[0] = d0_.x; b_[1] = d0_.y; b_[2] = d0_.z; b_[3] = d0_.w; \
       b_[4] = d1_.x; b_[5] = d1_.y; b_[6] = d1_.z; b_[7] = d1_.w; b_; })
  #define BIN_COMPUTE(KS, BUF)                                            \
    {                                                                     \
      const int kk0 = (KS) * 64 + lh * 16;                                \
      const uint2 c0 = *(const uint2*)(arow + kk0);                       \
      const uint2 c1 = *(const uint2*)(arow + kk0 + 8);                   \
      const uint2 c2 = *(const uint2*)(arow + kk0 + 32);                  \
      const uint2 c3 = *(const uint2*)(arow + kk0 + 40);                  \
      v8i a;                                                              \
      a[0] = (int)c0.x; a[1] = (int)c0.y; a[2] = (int)c1.x; a[3] = (int)c1.y; \
      a[4] = (int)c2.x; a[5] = (int)c2.y; a[6] = (int)c3.x; a[7] = (int)c3.y; \
      const unsigned char* p = &sB[BUF][0] + lm * 80 + lh * 16;           \
      const v8i b0 = BIN_LDB(p);                                          \
      const v8i b1 = BIN_LDB(p + 16 * 80);                                \
      const v8i b2 = BIN_LDB(p + 32 * 80);                                \
      const v8i b3 = BIN_LDB(p + 48 * 80);                                \
      acc0 = __builtin_amdgcn_wmma_i32_16x16x64_iu8(false, a, true, b0, acc0, false, false); \
      acc1 = __builtin_amdgcn_wmma_i32_16x16x64_iu8(false, a, true, b1, acc1, false, false); \
      acc2 = __builtin_amdgcn_wmma_i32_16x16x64_iu8(false, a, true, b2, acc2, false, false); \
      acc3 = __builtin_amdgcn_wmma_i32_16x16x64_iu8(false, a, true, b3, acc3, false, false); \
    }

  async_b128(ldsB0 + ldsRow,      gRow,      wsgn);
  async_b128(ldsB0 + ldsRow + 16, gRow + 16, wsgn);

  constexpr int NS = kHP / 64;
  for (int ks = 0; ks < NS; ks += 2) {
    wait_async0();
    __syncthreads();
    {
      const unsigned l = ldsB0 + 64u * 80u + ldsRow;
      const unsigned g = gRow + (unsigned)((ks + 1) * 64);
      async_b128(l,      g,      wsgn);
      async_b128(l + 16, g + 16, wsgn);
    }
    BIN_COMPUTE(ks, 0)
    wait_async0();
    __syncthreads();
    if (ks + 2 < NS) {
      const unsigned g = gRow + (unsigned)((ks + 2) * 64);
      async_b128(ldsB0 + ldsRow,      g,      wsgn);
      async_b128(ldsB0 + ldsRow + 16, g + 16, wsgn);
    }
    BIN_COMPUTE(ks + 1, 1)
  }
  #undef BIN_COMPUTE
  #undef BIN_LDB

  const float scale = (sumW[0] * invW) * (sumHin[0] * invH);
  const v8i accs[4] = {acc0, acc1, acc2, acc3};
  float lsum = 0.0f;
  #pragma unroll
  for (int t = 0; t < 4; ++t) {
    const int n = tileN + t * 16 + lm;
    const float bb = (n < kH) ? bias[n] : 0.0f;
    #pragma unroll
    for (int r = 0; r < 8; ++r) {
      const int m = tileM + lh * 8 + r;
      float v = scale * (float)accs[t][r] + bb;
      v = (v > 0.0f) ? v : 0.0f;
      if (bitsOut) {
        bitsOut[(size_t)m * kHP + n] = (n < kH && v > 0.0f) ? 1 : 0;
        if (n < kH) lsum += v;
      } else if (n < kH) {
        h4out[(size_t)m * kH + n] = v;
      }
    }
  }
  if (bitsOut) {
    for (int off = 16; off; off >>= 1) lsum += __shfl_down(lsum, off, 32);
    if (lane == 0) atomicAdd(sumHout, lsum);
  }
}

__global__ __launch_bounds__(256) void k_fc5(
    const float* __restrict__ h4, const float* __restrict__ w5,
    const float* __restrict__ b5, float* __restrict__ out)
{
  const int lane = threadIdx.x;
  const int m = blockIdx.x * blockDim.y + threadIdx.y;
  const float* hrow = h4 + (size_t)m * kH;
  float sc[kC];
  #pragma unroll
  for (int c = 0; c < kC; ++c) sc[c] = 0.0f;
  for (int k = lane; k < kH; k += 32) {
    const float hv = hrow[k];
    #pragma unroll
    for (int c = 0; c < kC; ++c) sc[c] += hv * w5[c * kH + k];
  }
  #pragma unroll
  for (int c = 0; c < kC; ++c) {
    float v = sc[c];
    for (int off = 16; off; off >>= 1) v += __shfl_down(v, off, 32);
    if (lane == 0) out[(size_t)m * kC + c] = v + b5[c];
  }
}

extern "C" void kernel_launch(void* const* d_in, const int* in_sizes, int n_in,
                              void* d_out, int out_size, void* d_ws, size_t ws_size,
                              hipStream_t stream)
{
  (void)in_sizes; (void)n_in; (void)out_size; (void)ws_size;

  const float* x  = (const float*)d_in[0];
  const float* W1 = (const float*)d_in[1];
  const float* b1 = (const float*)d_in[2];
  const float* W2 = (const float*)d_in[3];
  const float* b2 = (const float*)d_in[4];
  const float* W3 = (const float*)d_in[5];
  const float* b3 = (const float*)d_in[6];
  const float* W4 = (const float*)d_in[7];
  const float* b4 = (const float*)d_in[8];
  const float* W5 = (const float*)d_in[9];
  const float* b5 = (const float*)d_in[10];
  float* out = (float*)d_out;

  char* ws = (char*)d_ws;
  float*         scal  = (float*)(ws + OFF_SCAL);
  __bf16*        w1s   = (__bf16*)(ws + OFF_W1S);
  signed char*   ws2   = (signed char*)(ws + OFF_WS2);
  signed char*   ws3   = (signed char*)(ws + OFF_WS3);
  signed char*   ws4   = (signed char*)(ws + OFF_WS4);
  unsigned char* bits0 = (unsigned char*)(ws + OFF_BITS0);
  unsigned char* bits1 = (unsigned char*)(ws + OFF_BITS1);
  float*         h4    = (float*)(ws + OFF_H4);
  __bf16*        xbf   = (__bf16*)(ws + OFF_XBF);

  const float invWk = 1.0f / ((float)kH * (float)kH);
  const float invH  = 1.0f / ((float)kB * (float)kH);

  k_zero_scal<<<1, 32, 0, stream>>>(scal);
  k_abssum<<<512, 256, 0, stream>>>(W1, kH * kDin, scal + 0);
  k_abssum<<<512, 256, 0, stream>>>(W2, kH * kH,   scal + 1);
  k_abssum<<<512, 256, 0, stream>>>(W3, kH * kH,   scal + 2);
  k_abssum<<<512, 256, 0, stream>>>(W4, kH * kH,   scal + 3);
  k_x_bf16<<<4096, 256, 0, stream>>>(x, xbf);
  k_sign_bf16<<<4096, 256, 0, stream>>>(W1, w1s);
  k_sign_i8<<<4096, 256, 0, stream>>>(W2, ws2);
  k_sign_i8<<<4096, 256, 0, stream>>>(W3, ws3);
  k_sign_i8<<<4096, 256, 0, stream>>>(W4, ws4);

  const dim3 gg(kB / 64, kHP / 64);
  const dim3 bb(32, 4);

  k_gemm_l1<<<gg, bb, 0, stream>>>(xbf, w1s, b1, scal + 0, bits0, scal + 4);
  k_gemm_bin<<<gg, bb, 0, stream>>>(bits0, ws2, b2, scal + 1, invWk,
                                    scal + 4, invH, bits1, scal + 5, nullptr);
  k_gemm_bin<<<gg, bb, 0, stream>>>(bits1, ws3, b3, scal + 2, invWk,
                                    scal + 5, invH, bits0, scal + 6, nullptr);
  k_gemm_bin<<<gg, bb, 0, stream>>>(bits0, ws4, b4, scal + 3, invWk,
                                    scal + 6, invH, nullptr, nullptr, h4);
  k_fc5<<<kB / 8, dim3(32, 8), 0, stream>>>(h4, W5, b5, out);
}
